// InvariantMultiACE_6184752906369
// MI455X (gfx1250) — matrix operand order repaired and checked by reading, not compiled, not run on hardware
//
#include <hip/hip_runtime.h>
#include <hip/hip_bf16.h>
#include <math.h>

#define NN      10000
#define EE      160000
#define KCH     64
#define NELEM_  10
#define NB_     8
#define NG_     16
#define RMAXF   5.0f
#define AVGNEI  16.0f
#define SQ4PI   3.5449077018110318f
#define PI_F    3.14159265358979323846f

typedef __attribute__((ext_vector_type(16))) _Float16 half16;
typedef __attribute__((ext_vector_type(8)))  float    f32x8;

static __device__ __forceinline__ f32x8 wmma_f16f32(half16 a, half16 b, f32x8 c) {
  return __builtin_amdgcn_wmma_f32_16x16x32_f16(false, a, false, b, (short)0, c, false, false);
}

// Wave-local LDS fence: order this wave's ds_stores before subsequent ds_loads.
static __device__ __forceinline__ void lds_fence() {
  asm volatile("s_wait_dscnt 0x0" ::: "memory");
}

// Load a packed B fragment (one 32-byte vector per lane).
static __device__ __forceinline__ half16 ld_frag(const _Float16* p, int fragIdx) {
  int l = threadIdx.x & 31;
  return *(const half16*)(p + (((size_t)fragIdx * 32 + l) << 4));
}

static __device__ __forceinline__ void prefetch_frag(const _Float16* p, int fragIdx) {
  int l = threadIdx.x & 31;
  __builtin_prefetch((const void*)(p + (((size_t)fragIdx * 32 + l) << 4)), 0, 3);
}

// Build an A fragment (16x32, f16) from a row-major f32 matrix.
// ISA layout: lanes 0-15 -> M=l, elems 0..7 => K 0..7, 8..15 => K 16..23;
//             lanes 16-31 -> M=l-16, +8 K offset.
static __device__ __forceinline__ half16 afrag_f32(const float* A, int row0, int ld, int k0) {
  int l = threadIdx.x & 31;
  int m = row0 + (l & 15);
  int khi = (l & 16) ? 8 : 0;
  half16 a;
#pragma unroll
  for (int i = 0; i < 16; ++i) {
    int kk = k0 + khi + ((i < 8) ? i : (i + 8));
    a[i] = (_Float16)A[(size_t)m * ld + kk];
  }
  return a;
}

// ---------------- pack weights into B-fragment order (f16) ----------------
// dst index: (((kt*NT + nt)*32 + lane)*16 + i) ; B elem i -> K = kt*32 + (i<8 ? i : i+8) + 8*(lane>=16)
__global__ void k_pack_bfrag(const float* __restrict__ src, _Float16* __restrict__ dst,
                             int Kdim, int Ndim, int rs, int cs) {
  int t = blockIdx.x * 256 + threadIdx.x;
  if (t >= Kdim * Ndim) return;
  int i = t & 15;
  int l = (t >> 4) & 31;
  int tile = t >> 9;
  int NT = Ndim >> 4;
  int nt = tile % NT;
  int kt = tile / NT;
  int kk = kt * 32 + ((i < 8) ? i : (i + 8)) + ((l & 16) ? 8 : 0);
  int n  = nt * 16 + (l & 15);
  dst[t] = (_Float16)src[(size_t)kk * rs + (size_t)n * cs];
}

// ---------------- edge geometry: vec, r, sh(16), rb(8), drb/dr(8) ----------------
__global__ void k_geom(const float* __restrict__ pos, const float* __restrict__ shifts,
                       const int* __restrict__ eidx, float* __restrict__ vec,
                       float* __restrict__ rr, float* __restrict__ sh,
                       float* __restrict__ rb, float* __restrict__ drbdr) {
  int e = blockIdx.x * 256 + threadIdx.x;
  if (e >= EE) return;
  int s = eidx[e], rc = eidx[EE + e];
  float vx = pos[rc*3+0] - pos[s*3+0] + shifts[e*3+0];
  float vy = pos[rc*3+1] - pos[s*3+1] + shifts[e*3+1];
  float vz = pos[rc*3+2] - pos[s*3+2] + shifts[e*3+2];
  float r = sqrtf(vx*vx + vy*vy + vz*vz + 1e-12f);
  vec[e*3+0]=vx; vec[e*3+1]=vy; vec[e*3+2]=vz; rr[e]=r;
  float x=vx/r, y=vy/r, z=vz/r;
  float x2=x*x, y2=y*y, z2=z*z;
  float* o = sh + (size_t)e*16;
  o[0]  = SQ4PI*0.28209479177387814f;
  o[1]  = SQ4PI*0.4886025119029199f*y;
  o[2]  = SQ4PI*0.4886025119029199f*z;
  o[3]  = SQ4PI*0.4886025119029199f*x;
  o[4]  = SQ4PI*1.0925484305920792f*x*y;
  o[5]  = SQ4PI*1.0925484305920792f*y*z;
  o[6]  = SQ4PI*0.31539156525252005f*(3.f*z2-1.f);
  o[7]  = SQ4PI*1.0925484305920792f*x*z;
  o[8]  = SQ4PI*0.5462742152960396f*(x2-y2);
  o[9]  = SQ4PI*0.5900435899266435f*y*(3.f*x2-y2);
  o[10] = SQ4PI*2.890611442640554f*x*y*z;
  o[11] = SQ4PI*0.4570457994644658f*y*(5.f*z2-1.f);
  o[12] = SQ4PI*0.3731763325901154f*z*(5.f*z2-3.f);
  o[13] = SQ4PI*0.4570457994644658f*x*(5.f*z2-1.f);
  o[14] = SQ4PI*1.445305721320277f*z*(x2-y2);
  o[15] = SQ4PI*0.5900435899266435f*x*(x2-3.f*y2);
  // radial basis + its r-derivative
  float u = r / RMAXF;
  float fc = 0.f, fcp = 0.f;
  if (u < 1.f) {
    float u2=u*u, u4=u2*u2, u5=u4*u, u6=u5*u, u7=u6*u;
    fc  = 1.f - 21.f*u5 + 35.f*u6 - 15.f*u7;
    fcp = (-105.f*u4 + 210.f*u5 - 105.f*u6) / RMAXF;
  }
  const float pref = 0.6324555320336759f; // sqrt(2/R_MAX)
#pragma unroll
  for (int b = 0; b < NB_; ++b) {
    float w = (float)(b+1) * PI_F / RMAXF;
    float sn = __sinf(w*r), cn = __cosf(w*r);
    rb[e*NB_+b]    = pref * sn / r * fc;
    drbdr[e*NB_+b] = pref * ((w*cn/r - sn/(r*r))*fc + sn/r*fcp);
  }
}

// ---------------- h0 = node_attrs @ embed_W ----------------
__global__ void k_embed(const float* __restrict__ attrs, const float* __restrict__ eW,
                        float* __restrict__ h0) {
  int t = blockIdx.x * 256 + threadIdx.x;
  if (t >= NN * KCH) return;
  int n = t >> 6, k = t & 63;
  float acc = 0.f;
#pragma unroll
  for (int e = 0; e < NELEM_; ++e) acc += attrs[n*NELEM_+e] * eW[e*KCH+k];
  h0[t] = acc;
}

// ---------------- z = rb@Wr1 ; t = silu(z) stored packed in A-fragment order ----------------
// Each thread handles an adjacent feature pair (same fragment half-pair -> merged stores).
__global__ void k_radial(const float* __restrict__ rb, const float* __restrict__ Wr1l,
                         float* __restrict__ zbuf, _Float16* __restrict__ tpack) {
  int t = blockIdx.x * 256 + threadIdx.x;
  if (t >= EE * 32) return;
  int e = t >> 5;
  int f = (t & 31) * 2;
  float a0 = 0.f, a1 = 0.f;
#pragma unroll
  for (int b = 0; b < NB_; ++b) {
    float rv = rb[e*NB_+b];
    a0 += rv * Wr1l[b*KCH + f];
    a1 += rv * Wr1l[b*KCH + f + 1];
  }
  zbuf[(size_t)e*64 + f]     = a0;
  zbuf[(size_t)e*64 + f + 1] = a1;
  float s0 = 1.f / (1.f + __expf(-a0));
  float s1 = 1.f / (1.f + __expf(-a1));
  float t0 = a0 * s0, t1 = a1 * s1;
  int eblk = e >> 4, m = e & 15, ks = f >> 5, fr = f & 31;
  int half_sel = (fr >> 3) & 1;
  int lane = m + 16 * half_sel;
  int i = (fr & 7) + ((fr & 16) ? 8 : 0);
  size_t base = ((((size_t)eblk*2 + ks)*32 + lane) << 4) + i;
  tpack[base]   = (_Float16)t0;
  tpack[base+1] = (_Float16)t1;
}

// ---------------- WMMA GEMM: C(N x 64) = A(N x 64) @ Bpack(64 x 64) ----------------
__global__ __launch_bounds__(32) void k_gemm64(const float* __restrict__ A,
                                               const _Float16* __restrict__ Bpack,
                                               float* __restrict__ C) {
  int rt = blockIdx.x;
  int l = threadIdx.x;
  int row0 = rt * 16;
  half16 a0 = afrag_f32(A, row0, 64, 0);
  half16 a1 = afrag_f32(A, row0, 64, 32);
  int mrow = 8 * (l >> 4), j = l & 15;
#pragma unroll
  for (int nt = 0; nt < 4; ++nt) {
    f32x8 c = {};
    c = wmma_f16f32(a0, ld_frag(Bpack, 0*4 + nt), c);
    c = wmma_f16f32(a1, ld_frag(Bpack, 1*4 + nt), c);
#pragma unroll
    for (int r_ = 0; r_ < 8; ++r_)
      C[(size_t)(row0 + mrow + r_)*64 + nt*16 + j] = c[r_];
  }
}

// ---------------- fused forward message: R tiles -> msg -> atomic scatter into A ----------------
// 128 threads = 4 waves; each wave owns an independent 16-edge block.
__global__ __launch_bounds__(128) void k_msg_fwd(const _Float16* __restrict__ tpack,
                                                 const _Float16* __restrict__ wr2p,
                                                 const float* __restrict__ sh,
                                                 const float* __restrict__ hu,
                                                 const int* __restrict__ eidx,
                                                 float* __restrict__ Aout) {
  __shared__ float shs[4][256];
  __shared__ float hus[4][1024];
  __shared__ int   rcv[4][16];
  int w = threadIdx.x >> 5, l = threadIdx.x & 31;
  int eblk = blockIdx.x * 4 + w;
  int e0 = eblk * 16;
  float* shsw = shs[w];
  float* husw = hus[w];
  int*   rcvw = rcv[w];
  if (l < 16) rcvw[l] = eidx[EE + e0 + l];
  for (int idx = l; idx < 256; idx += 32) shsw[idx] = sh[(size_t)e0*16 + idx];
  for (int idx = l; idx < 1024; idx += 32) {
    int m = idx >> 6;
    husw[idx] = hu[(size_t)eidx[e0+m]*64 + (idx & 63)];
  }
  lds_fence();
  half16 a0 = *(const half16*)(tpack + ((((size_t)eblk*2 + 0)*32 + l) << 4));
  half16 a1 = *(const half16*)(tpack + ((((size_t)eblk*2 + 1)*32 + l) << 4));
  int mrow = 8 * (l >> 4), j = l & 15;
  for (int nt = 0; nt < 64; ++nt) {
    if (nt < 63) {
      prefetch_frag(wr2p, 0*64 + nt + 1);
      prefetch_frag(wr2p, 1*64 + nt + 1);
    }
    f32x8 c = {};
    c = wmma_f16f32(a0, ld_frag(wr2p, 0*64 + nt), c);
    c = wmma_f16f32(a1, ld_frag(wr2p, 1*64 + nt), c);
#pragma unroll
    for (int r_ = 0; r_ < 8; ++r_) {
      int m = mrow + r_;
      float v = c[r_] * shsw[m*16 + j] * husw[m*64 + nt] * (1.f/AVGNEI);
      atomicAdd(&Aout[(size_t)rcvw[m]*1024 + nt*16 + j], v);
    }
  }
}

// ---------------- product basis forward ----------------
__global__ void k_pb_fwd(const float* __restrict__ A, const float* __restrict__ Wpl,
                         float* __restrict__ hout) {
  int t = blockIdx.x * 256 + threadIdx.x;
  if (t >= NN * KCH) return;
  int n = t >> 6, k = t & 63;
  const float* a = A + (size_t)n*1024 + k*16;
  float A0 = a[0];
  float p0 = A0*A0;
  float p1 = a[1]*a[1] + a[2]*a[2] + a[3]*a[3];
  float p2 = 0.f, p3 = 0.f;
#pragma unroll
  for (int m = 4; m < 9; ++m)  p2 += a[m]*a[m];
#pragma unroll
  for (int m = 9; m < 16; ++m) p3 += a[m]*a[m];
  float P = p0 + p1 + p2 + p3;
  const float* w = Wpl + k*6;
  hout[t] = A0*w[0] + p0*w[1] + p1*w[2] + p2*w[3] + p3*w[4] + A0*P*w[5];
}

// ---------------- product basis backward: dh -> dA ----------------
__global__ void k_pb_bwd(const float* __restrict__ A, const float* __restrict__ dh,
                         const float* __restrict__ Wpl, float* __restrict__ dA) {
  int t = blockIdx.x * 256 + threadIdx.x;
  if (t >= NN * KCH) return;
  int n = t >> 6, k = t & 63;
  const float* a = A + (size_t)n*1024 + k*16;
  float* da = dA + (size_t)n*1024 + k*16;
  const float* w = Wpl + k*6;
  float g = dh[t];
  float A0 = a[0];
  float P = 0.f;
#pragma unroll
  for (int m = 0; m < 16; ++m) P += a[m]*a[m];
#pragma unroll
  for (int m = 0; m < 16; ++m) {
    int lm = (m == 0) ? 0 : (m < 4) ? 1 : (m < 9) ? 2 : 3;
    float v = 2.f * a[m] * (w[1+lm] + w[5]*A0);
    if (m == 0) v += w[0] + w[5]*P;
    da[m] = g * v;
  }
}

// ---------------- readout + e0 -> energies ----------------
__global__ void k_readout(const float* __restrict__ h2, const float* __restrict__ Wr,
                          const float* __restrict__ attrs, const float* __restrict__ ae,
                          const int* __restrict__ batch, float* __restrict__ out) {
  int n = blockIdx.x * 256 + threadIdx.x;
  if (n >= NN) return;
  float ne = 0.f;
#pragma unroll
  for (int k = 0; k < KCH; ++k) ne += h2[(size_t)n*64 + k] * Wr[k];
  float e0 = 0.f;
#pragma unroll
  for (int e = 0; e < NELEM_; ++e) e0 += attrs[n*NELEM_+e] * ae[e];
  atomicAdd(&out[batch[n]], ne + e0);
}

__global__ void k_dhinit(const float* __restrict__ Wr, float* __restrict__ dh) {
  int t = blockIdx.x * 256 + threadIdx.x;
  if (t >= NN * KCH) return;
  dh[t] = Wr[t & 63];
}

// ---------------- fused backward message kernel (4 waves / block) ----------------
__global__ __launch_bounds__(128) void k_msg_bwd(
    const _Float16* __restrict__ tpack, const _Float16* __restrict__ wr2p,
    const _Float16* __restrict__ wr2tp, const float* __restrict__ dA,
    const float* __restrict__ sh, const float* __restrict__ hu,
    const float* __restrict__ zbuf, const float* __restrict__ Wr1l,
    const float* __restrict__ drbdr, const float* __restrict__ vec,
    const float* __restrict__ rr, const int* __restrict__ eidx,
    float* __restrict__ dhu, float* __restrict__ gpos, int accum_dhu) {
  __shared__ float shs[4][256];
  __shared__ float hus[4][1024];
  __shared__ int   rcv[4][16];
  __shared__ int   snd[4][16];
  __shared__ _Float16 slab[4][16*32];
  __shared__ float dzs[4][16*64];
  __shared__ float dss[4][256];
  int w = threadIdx.x >> 5, l = threadIdx.x & 31;
  int eblk = blockIdx.x * 4 + w;
  int e0 = eblk * 16;
  float* shsw = shs[w];
  float* husw = hus[w];
  int*   rcvw = rcv[w];
  int*   sndw = snd[w];
  _Float16* slabw = slab[w];
  float* dzsw = dzs[w];
  float* dssw = dss[w];
  if (l < 16) { rcvw[l] = eidx[EE + e0 + l]; sndw[l] = eidx[e0 + l]; }
  for (int idx = l; idx < 256; idx += 32) shsw[idx] = sh[(size_t)e0*16 + idx];
  for (int idx = l; idx < 1024; idx += 32) {
    int m = idx >> 6;
    husw[idx] = hu[(size_t)eidx[e0+m]*64 + (idx & 63)];
  }
  lds_fence();
  half16 a0 = *(const half16*)(tpack + ((((size_t)eblk*2 + 0)*32 + l) << 4));
  half16 a1 = *(const half16*)(tpack + ((((size_t)eblk*2 + 1)*32 + l) << 4));
  int mrow = 8 * (l >> 4), j = l & 15;
  float dsa[8];
#pragma unroll
  for (int r_ = 0; r_ < 8; ++r_) dsa[r_] = 0.f;
  f32x8 dtc[4];
#pragma unroll
  for (int ft = 0; ft < 4; ++ft) dtc[ft] = (f32x8){};

  for (int k2 = 0; k2 < 32; ++k2) {
    if (k2 < 31) {
      prefetch_frag(wr2p,  0*64 + 2*k2 + 2);
      prefetch_frag(wr2p,  1*64 + 2*k2 + 2);
      prefetch_frag(wr2tp, (k2+1)*4);
    }
#pragma unroll
    for (int sub = 0; sub < 2; ++sub) {
      int nt = 2*k2 + sub;
      f32x8 c = {};
      c = wmma_f16f32(a0, ld_frag(wr2p, 0*64 + nt), c);
      c = wmma_f16f32(a1, ld_frag(wr2p, 1*64 + nt), c);
#pragma unroll
      for (int r_ = 0; r_ < 8; ++r_) {
        int m = mrow + r_;
        float dAv = dA[(size_t)rcvw[m]*1024 + nt*16 + j] * (1.f/AVGNEI);
        float Rv  = c[r_];
        float shv = shsw[m*16 + j];
        float huv = husw[m*64 + nt];
        float dR  = dAv * shv * huv;
        dsa[r_]  += dAv * Rv * huv;
        float ph  = dAv * Rv * shv;
        ph += __shfl_xor(ph, 1);
        ph += __shfl_xor(ph, 2);
        ph += __shfl_xor(ph, 4);
        ph += __shfl_xor(ph, 8);
        if (accum_dhu && j == 0) atomicAdd(&dhu[(size_t)sndw[m]*64 + nt], ph);
        slabw[m*32 + sub*16 + j] = (_Float16)dR;
      }
    }
    lds_fence();
    half16 adr;
    {
      int khi = (l & 16) ? 8 : 0;
      int m = l & 15;
#pragma unroll
      for (int i = 0; i < 16; ++i) {
        int kk = khi + ((i < 8) ? i : (i + 8));
        adr[i] = slabw[m*32 + kk];
      }
    }
#pragma unroll
    for (int ft = 0; ft < 4; ++ft)
      dtc[ft] = wmma_f16f32(adr, ld_frag(wr2tp, k2*4 + ft), dtc[ft]);
    lds_fence();
  }

  // dz = dt * silu'(z) -> LDS ; dsh -> LDS
#pragma unroll
  for (int ft = 0; ft < 4; ++ft) {
#pragma unroll
    for (int r_ = 0; r_ < 8; ++r_) {
      int m = mrow + r_;
      int f = ft*16 + j;
      float zv = zbuf[(size_t)(e0 + m)*64 + f];
      float s = 1.f / (1.f + __expf(-zv));
      dzsw[m*64 + f] = dtc[ft][r_] * s * (1.f + zv*(1.f - s));
    }
  }
#pragma unroll
  for (int r_ = 0; r_ < 8; ++r_) dssw[(mrow + r_)*16 + j] = dsa[r_];
  lds_fence();

  if (l < 16) {
    int el = l, e = e0 + l;
    // drb = dz @ Wr1^T ; dr_tot = drb . drb/dr
    float drtot = 0.f;
#pragma unroll
    for (int b = 0; b < NB_; ++b) {
      float s = 0.f;
      for (int f = 0; f < 64; ++f) s += dzsw[el*64 + f] * Wr1l[b*64 + f];
      drtot += s * drbdr[e*NB_ + b];
    }
    float vx = vec[e*3+0], vy = vec[e*3+1], vz = vec[e*3+2];
    float r = rr[e];
    float x = vx/r, y = vy/r, uz = vz/r;
    float x2 = x*x, y2 = y*y, z2 = uz*uz;
    float q[16];
#pragma unroll
    for (int jj = 0; jj < 16; ++jj) q[jj] = dssw[el*16 + jj] * SQ4PI;
    const float c1 = 0.4886025119029199f, c2 = 1.0925484305920792f;
    const float c6 = 0.31539156525252005f, c8 = 0.5462742152960396f;
    const float c9 = 0.5900435899266435f, c10 = 2.890611442640554f;
    const float c11 = 0.4570457994644658f, c12 = 0.3731763325901154f;
    const float c14 = 1.445305721320277f;
    float gx = 0.f, gy = 0.f, gz = 0.f, d;
    d = q[1];  gy += c1*d;
    d = q[2];  gz += c1*d;
    d = q[3];  gx += c1*d;
    d = q[4];  gx += c2*y*d;             gy += c2*x*d;
    d = q[5];  gy += c2*uz*d;            gz += c2*y*d;
    d = q[6];  gz += 6.f*c6*uz*d;
    d = q[7];  gx += c2*uz*d;            gz += c2*x*d;
    d = q[8];  gx += 2.f*c8*x*d;         gy -= 2.f*c8*y*d;
    d = q[9];  gx += 6.f*c9*x*y*d;       gy += 3.f*c9*(x2 - y2)*d;
    d = q[10]; gx += c10*y*uz*d;         gy += c10*x*uz*d;       gz += c10*x*y*d;
    d = q[11]; gy += c11*(5.f*z2-1.f)*d; gz += 10.f*c11*y*uz*d;
    d = q[12]; gz += c12*(15.f*z2-3.f)*d;
    d = q[13]; gx += c11*(5.f*z2-1.f)*d; gz += 10.f*c11*x*uz*d;
    d = q[14]; gx += 2.f*c14*x*uz*d;     gy -= 2.f*c14*y*uz*d;   gz += c14*(x2-y2)*d;
    d = q[15]; gx += 3.f*c9*(x2-y2)*d;   gy -= 6.f*c9*x*y*d;
    float udot = x*gx + y*gy + uz*gz;
    float dvx = (gx - x*udot)/r + drtot*x;
    float dvy = (gy - y*udot)/r + drtot*y;
    float dvz = (gz - uz*udot)/r + drtot*uz;
    int rc = rcvw[el], sn = sndw[el];
    atomicAdd(&gpos[rc*3+0],  dvx);
    atomicAdd(&gpos[rc*3+1],  dvy);
    atomicAdd(&gpos[rc*3+2],  dvz);
    atomicAdd(&gpos[sn*3+0], -dvx);
    atomicAdd(&gpos[sn*3+1], -dvy);
    atomicAdd(&gpos[sn*3+2], -dvz);
  }
}

__global__ void k_finalize(const float* __restrict__ gpos, float* __restrict__ out) {
  int t = blockIdx.x * 256 + threadIdx.x;
  if (t >= NN * 3) return;
  out[NG_ + t] = -gpos[t];
}

extern "C" void kernel_launch(void* const* d_in, const int* in_sizes, int n_in,
                              void* d_out, int out_size, void* d_ws, size_t ws_size,
                              hipStream_t stream) {
  (void)in_sizes; (void)n_in; (void)out_size; (void)ws_size;
  const float* positions = (const float*)d_in[0];
  const float* node_attrs = (const float*)d_in[1];
  const float* shifts = (const float*)d_in[2];
  const int*   edge_index = (const int*)d_in[3];
  const int*   batch = (const int*)d_in[4];
  const float* atomic_energies = (const float*)d_in[6];
  const float* embed_W = (const float*)d_in[7];
  const float* Wr1 = (const float*)d_in[8];
  const float* Wr2 = (const float*)d_in[9];
  const float* lin_up = (const float*)d_in[10];
  const float* Wp = (const float*)d_in[11];
  const float* W_read = (const float*)d_in[12];
  float* out = (float*)d_out;

  char* ws = (char*)d_ws;
  size_t cur = 0;
  auto carve = [&](size_t bytes) -> void* {
    void* p = ws + cur;
    cur = (cur + bytes + 255) & ~(size_t)255;
    return p;
  };
  float* vecb   = (float*)carve((size_t)EE*3*4);
  float* rrb    = (float*)carve((size_t)EE*4);
  float* shb    = (float*)carve((size_t)EE*16*4);
  float* rbb    = (float*)carve((size_t)EE*NB_*4);
  float* drbb   = (float*)carve((size_t)EE*NB_*4);
  float* zb0    = (float*)carve((size_t)EE*64*4);
  float* zb1    = (float*)carve((size_t)EE*64*4);
  float* Ab0    = (float*)carve((size_t)NN*1024*4);
  float* Ab1    = (float*)carve((size_t)NN*1024*4);
  float* h0     = (float*)carve((size_t)NN*64*4);
  float* h1     = (float*)carve((size_t)NN*64*4);
  float* h2     = (float*)carve((size_t)NN*64*4);
  float* hu0    = (float*)carve((size_t)NN*64*4);
  float* hu1    = (float*)carve((size_t)NN*64*4);
  float* dhb    = (float*)carve((size_t)NN*64*4);
  float* dhub   = (float*)carve((size_t)NN*64*4);
  float* dAb    = (float*)carve((size_t)NN*1024*4);
  float* gposb  = (float*)carve((size_t)NN*3*4);
  _Float16* tp0   = (_Float16*)carve((size_t)EE*64*2);
  _Float16* tp1   = (_Float16*)carve((size_t)EE*64*2);
  _Float16* wr2p0 = (_Float16*)carve((size_t)64*1024*2);
  _Float16* wr2p1 = (_Float16*)carve((size_t)64*1024*2);
  _Float16* wr2t0 = (_Float16*)carve((size_t)64*1024*2);
  _Float16* wr2t1 = (_Float16*)carve((size_t)64*1024*2);
  _Float16* lup0  = (_Float16*)carve((size_t)64*64*2);
  _Float16* lup1  = (_Float16*)carve((size_t)64*64*2);
  _Float16* lupT1 = (_Float16*)carve((size_t)64*64*2);

  hipMemsetAsync(Ab0,   0, (size_t)NN*1024*4, stream);
  hipMemsetAsync(Ab1,   0, (size_t)NN*1024*4, stream);
  hipMemsetAsync(gposb, 0, (size_t)NN*3*4, stream);
  hipMemsetAsync(dhub,  0, (size_t)NN*64*4, stream);
  hipMemsetAsync(out,   0, (size_t)NG_*4, stream);

  // weight packing
  k_pack_bfrag<<<256, 256, 0, stream>>>(Wr2,          wr2p0, 64, 1024, 1024, 1);
  k_pack_bfrag<<<256, 256, 0, stream>>>(Wr2 + 65536,  wr2p1, 64, 1024, 1024, 1);
  k_pack_bfrag<<<256, 256, 0, stream>>>(Wr2,          wr2t0, 1024, 64, 1, 1024);
  k_pack_bfrag<<<256, 256, 0, stream>>>(Wr2 + 65536,  wr2t1, 1024, 64, 1, 1024);
  k_pack_bfrag<<<16, 256, 0, stream>>>(lin_up,        lup0,  64, 64, 64, 1);
  k_pack_bfrag<<<16, 256, 0, stream>>>(lin_up + 4096, lup1,  64, 64, 64, 1);
  k_pack_bfrag<<<16, 256, 0, stream>>>(lin_up + 4096, lupT1, 64, 64, 1, 64);

  // forward
  k_geom<<<(EE + 255)/256, 256, 0, stream>>>(positions, shifts, edge_index,
                                             vecb, rrb, shb, rbb, drbb);
  k_embed<<<(NN*KCH + 255)/256, 256, 0, stream>>>(node_attrs, embed_W, h0);

  k_gemm64<<<NN/16, 32, 0, stream>>>(h0, lup0, hu0);
  k_radial<<<(EE*32 + 255)/256, 256, 0, stream>>>(rbb, Wr1, zb0, tp0);
  k_msg_fwd<<<EE/64, 128, 0, stream>>>(tp0, wr2p0, shb, hu0, edge_index, Ab0);
  k_pb_fwd<<<(NN*KCH + 255)/256, 256, 0, stream>>>(Ab0, Wp, h1);

  k_gemm64<<<NN/16, 32, 0, stream>>>(h1, lup1, hu1);
  k_radial<<<(EE*32 + 255)/256, 256, 0, stream>>>(rbb, Wr1 + 512, zb1, tp1);
  k_msg_fwd<<<EE/64, 128, 0, stream>>>(tp1, wr2p1, shb, hu1, edge_index, Ab1);
  k_pb_fwd<<<(NN*KCH + 255)/256, 256, 0, stream>>>(Ab1, Wp + 384, h2);

  k_readout<<<(NN + 255)/256, 256, 0, stream>>>(h2, W_read, node_attrs,
                                                atomic_energies, batch, out);

  // backward (forces)
  k_dhinit<<<(NN*KCH + 255)/256, 256, 0, stream>>>(W_read, dhb);
  k_pb_bwd<<<(NN*KCH + 255)/256, 256, 0, stream>>>(Ab1, dhb, Wp + 384, dAb);
  k_msg_bwd<<<EE/64, 128, 0, stream>>>(tp1, wr2p1, wr2t1, dAb, shb, hu1, zb1,
                                       Wr1 + 512, drbb, vecb, rrb, edge_index,
                                       dhub, gposb, 1);
  k_gemm64<<<NN/16, 32, 0, stream>>>(dhub, lupT1, dhb);
  k_pb_bwd<<<(NN*KCH + 255)/256, 256, 0, stream>>>(Ab0, dhb, Wp, dAb);
  k_msg_bwd<<<EE/64, 128, 0, stream>>>(tp0, wr2p0, wr2t0, dAb, shb, hu0, zb0,
                                       Wr1, drbb, vecb, rrb, edge_index,
                                       dhub, gposb, 0);
  k_finalize<<<(NN*3 + 255)/256, 256, 0, stream>>>(gposb, out);
}
